// TransformerBlock_2705829396538
// MI455X (gfx1250) — matrix operand order repaired and checked by reading, not compiled, not run on hardware
//
#include <hip/hip_runtime.h>

// ---------------------------------------------------------------------------
// Restormer-style transformer block for MI455X (gfx1250, wave32, WMMA).
// Intermediates in NHWC bf16; 1x1 convs = row-major GEMMs on
// v_wmma_f32_16x16x32_bf16 with fp32 accumulation. A-panels staged in LDS
// via GLOBAL_LOAD_ASYNC_TO_LDS_B128 (fallback: plain LDS stores).
// ---------------------------------------------------------------------------

#define NPX    131072   // B*H*W = 8*128*128
#define CCH    128
#define IMGHW  16384    // 128*128
#define NB     8
#define NHEADS 8

#if defined(__has_builtin)
#if __has_builtin(__builtin_amdgcn_global_load_async_to_lds_b128)
#define HAVE_ASYNC_LDS 1
#endif
#endif

typedef __attribute__((ext_vector_type(16))) __bf16 v16bf;
typedef __attribute__((ext_vector_type(8)))  float  v8f;
typedef __attribute__((ext_vector_type(4)))  int    v4i;

__device__ __forceinline__ unsigned short f2bf(float f) {
  unsigned int u = __float_as_uint(f);
  u += 0x7FFFu + ((u >> 16) & 1u);           // round-to-nearest-even
  return (unsigned short)(u >> 16);
}
__device__ __forceinline__ float bf2f(unsigned short h) {
  return __uint_as_float(((unsigned int)h) << 16);
}

// 16B global -> LDS copy, async path when available.
__device__ __forceinline__ void copy16_g2l(const unsigned short* src,
                                           unsigned short* lds_dst) {
#if defined(HAVE_ASYNC_LDS)
  __builtin_amdgcn_global_load_async_to_lds_b128(
      (__attribute__((address_space(1))) v4i*)(void*)src,
      (__attribute__((address_space(3))) v4i*)(void*)lds_dst,
      0, 0);
#else
  *(uint4*)lds_dst = *(const uint4*)src;
#endif
}
__device__ __forceinline__ void wait_async_lds() {
#if defined(HAVE_ASYNC_LDS)
  asm volatile("s_wait_asynccnt 0x0" ::: "memory");
#endif
}

union AF { unsigned int  u[8];  v16bf v; };  // fragment via dword loads
union SF { unsigned short s[16]; v16bf v; }; // fragment via b16 loads

// ---------------------------------------------------------------------------
// Pack fp32 weight [N,K] (O,I row-major) into WMMA B-fragment tile layout:
// tile (nt,kt): 32 lanes x 16 bf16 contiguous; lane=(half,n), elem e -> k=e+8*half (+8 if e>=8)
// ---------------------------------------------------------------------------
__global__ void pack_b(const float* __restrict__ W, unsigned short* __restrict__ out,
                       int N, int K) {
  const int total = N * K;
  const int ktiles = K >> 5;
  for (int idx = blockIdx.x * blockDim.x + threadIdx.x; idx < total;
       idx += gridDim.x * blockDim.x) {
    const int e    = idx & 15;
    const int lane = (idx >> 4) & 31;
    const int t    = idx >> 9;
    const int kt   = t % ktiles;
    const int nt   = t / ktiles;
    const int n = nt * 16 + (lane & 15);
    const int k = kt * 32 + e + 8 * (lane >> 4) + ((e >= 8) ? 8 : 0);
    out[idx] = f2bf(W[(size_t)n * K + k]);
  }
}

__global__ void zero_f32(float* __restrict__ p, int n) {
  int i = blockIdx.x * blockDim.x + threadIdx.x;
  if (i < n) p[i] = 0.f;
}

// ---------------------------------------------------------------------------
// LayerNorm over channels of NCHW fp32 input -> NHWC bf16 rows.
// ---------------------------------------------------------------------------
__global__ void ln1_k(const float* __restrict__ x, const float* __restrict__ g,
                      const float* __restrict__ b_, unsigned short* __restrict__ A1) {
  const int px  = blockIdx.x * blockDim.x + threadIdx.x;
  const int img = px >> 14;
  const int hw  = px & (IMGHW - 1);
  const float* base = x + ((size_t)img * CCH << 14) + hw;
  float s = 0.f, s2 = 0.f;
  for (int c = 0; c < CCH; ++c) {
    float v = base[(size_t)c << 14];
    s += v; s2 += v * v;
  }
  const float mu  = s * (1.f / 128.f);
  const float var = s2 * (1.f / 128.f) - mu * mu;
  const float rs  = rsqrtf(var + 1e-5f);
  unsigned short* o = A1 + (size_t)px * CCH;
  for (int c = 0; c < CCH; ++c) {
    float v = base[(size_t)c << 14];
    o[c] = f2bf((v - mu) * rs * g[c] + b_[c]);
  }
}

// ---------------------------------------------------------------------------
// bf16 WMMA GEMM: D[M,N] = A[M,K] * Bpacked + epilogue.
// 8 waves/block share one 16-row A panel (staged in LDS, +16B row pad to
// avoid bank conflicts); each wave owns one 16x16 tile. Grids must cover
// exactly (M/16)*(N/16) tiles with N/16 a multiple of 8.
// MODE 0: store bf16 [M,N].  MODE 1: xa = x_nchw + acc, store f32 NHWC.
// MODE 2: out_nchw = xa_nhwc + acc (final transpose store).
// ---------------------------------------------------------------------------
template <int MODE>
__global__ void gemm_bf16_wmma(const unsigned short* __restrict__ A,
                               const unsigned short* __restrict__ Bp,
                               int M, int N, int K,
                               unsigned short* __restrict__ outB,
                               float* __restrict__ outF,
                               const float* __restrict__ res) {
  __shared__ unsigned short aTile[16 * (512 + 8)];   // max K=512, padded rows
  const int wid  = threadIdx.x >> 5;
  const int lane = threadIdx.x & 31;
  const int half = lane >> 4;
  const int lr   = lane & 15;
  const int ntiles = N >> 4;
  const int ktiles = K >> 5;
  const int KP = K + 8;                              // padded LDS row stride
  const long tile = (long)blockIdx.x * 8 + wid;
  const int nt = (int)(tile % ntiles);
  const int mt = (int)(tile / ntiles);               // uniform across block

  // Cooperative A-panel stage: 16 rows x K bf16, 16B chunks.
  {
    const int row8 = K >> 3;                         // 16B chunks per row
    for (int i = threadIdx.x; i < 16 * row8; i += 256) {
      const int r = i / row8, c8 = i - r * row8;
      copy16_g2l(A + (size_t)(mt * 16 + r) * K + c8 * 8,
                 &aTile[r * KP + c8 * 8]);
    }
    wait_async_lds();
  }
  __syncthreads();

  const unsigned int* bbase =
      (const unsigned int*)Bp + ((size_t)nt * ktiles * 32 + lane) * 8;
  const unsigned short* arow = &aTile[lr * KP];

  v8f acc = {};
  for (int kt = 0; kt < ktiles; ++kt) {
    AF a;
    const int kb = kt * 32 + 8 * half;
#pragma unroll
    for (int j = 0; j < 8; ++j) {
      const int kk = kb + ((j < 4) ? (2 * j) : (16 + 2 * (j - 4)));
      a.u[j] = *(const unsigned int*)(arow + kk);
    }
    AF b;
    const unsigned int* bp = bbase + (size_t)kt * 32 * 8;
#pragma unroll
    for (int j = 0; j < 8; ++j) b.u[j] = bp[j];
    acc = __builtin_amdgcn_wmma_f32_16x16x32_bf16(false, a.v, false, b.v,
                                                  (short)0, acc, false, false);
  }

#pragma unroll
  for (int r = 0; r < 8; ++r) {
    const int row = mt * 16 + r + 8 * half;   // pixel
    const int col = nt * 16 + lr;             // channel
    if (MODE == 0) {
      outB[(size_t)row * N + col] = f2bf(acc[r]);
    } else if (MODE == 1) {
      const int img = row >> 14, hw = row & (IMGHW - 1);
      outF[(size_t)row * CCH + col] =
          res[(((size_t)img * CCH + col) << 14) + hw] + acc[r];
    } else {
      const int img = row >> 14, hw = row & (IMGHW - 1);
      outF[(((size_t)img * CCH + col) << 14) + hw] =
          res[(size_t)row * CCH + col] + acc[r];
    }
  }
}

// ---------------------------------------------------------------------------
// Depthwise 3x3 over NHWC bf16 [NPX,384], zero pad 1. blockDim = 384.
// ---------------------------------------------------------------------------
__global__ void dwconv3_384(const unsigned short* __restrict__ in,
                            const float* __restrict__ w,
                            unsigned short* __restrict__ out) {
  const int px = blockIdx.x, c = threadIdx.x;
  const int hh = (px >> 7) & 127, ww = px & 127;
  const float* wc = w + c * 9;
  float acc = 0.f;
  for (int dy = -1; dy <= 1; ++dy) {
    const int y = hh + dy; if ((unsigned)y >= 128u) continue;
    for (int dx = -1; dx <= 1; ++dx) {
      const int xx = ww + dx; if ((unsigned)xx >= 128u) continue;
      acc += bf2f(in[(size_t)(px + dy * 128 + dx) * 384 + c]) *
             wc[(dy + 1) * 3 + dx + 1];
    }
  }
  out[(size_t)px * 384 + c] = f2bf(acc);
}

// Sum of squares over spatial for q/k channels (0..255) per image.
__global__ void qk_norm(const unsigned short* __restrict__ qkv,
                        float* __restrict__ nrm) {
  const int img = blockIdx.x >> 4, strip = blockIdx.x & 15;
  const int c = threadIdx.x;                 // 0..255
  const size_t base = ((size_t)img * IMGHW + strip * 1024) * 384 + c;
  float s = 0.f;
  for (int i = 0; i < 1024; ++i) {
    float v = bf2f(qkv[base + (size_t)i * 384]);
    s += v * v;
  }
  atomicAdd(&nrm[img * 256 + c], s);
}

// ---------------------------------------------------------------------------
// Channel attention Gram matrix: G[b,h] = Q[b,h] * K[b,h]^T, K-dim = 16384.
// One block per (b,h); 8 waves split the reduction; per 32-wide K step each
// lane fetches its 32B q/k row slice as b128 into per-wave LDS tiles, then
// WMMA fragments are gathered from LDS.
// ---------------------------------------------------------------------------
__global__ void attn_qk_wmma(const unsigned short* __restrict__ qkv,
                             float* __restrict__ attnG) {
  __shared__ unsigned short qt[8][512];   // [wave][32 rows x 16 ch]
  __shared__ unsigned short kt2[8][512];
  const int bh = blockIdx.x;
  const int img = bh >> 3, h = bh & 7;
  const int wid = threadIdx.x >> 5, lane = threadIdx.x & 31;
  const int half = lane >> 4, lr = lane & 15;
  const size_t base = (size_t)img * IMGHW * 384;
  v8f acc = {};
  for (int it = 0; it < 64; ++it) {
    const int n0 = wid * 2048 + it * 32;
    // Stage: row n0+lane, 16 q channels and 16 k channels (32B each).
    const unsigned short* qrow = qkv + base + (size_t)(n0 + lane) * 384 + h * 16;
    copy16_g2l(qrow,       &qt[wid][lane * 16]);
    copy16_g2l(qrow + 8,   &qt[wid][lane * 16 + 8]);
    copy16_g2l(qrow + 128, &kt2[wid][lane * 16]);       // k = q + 128 ch
    copy16_g2l(qrow + 136, &kt2[wid][lane * 16 + 8]);
    wait_async_lds();
    __syncthreads();
    SF a, b;
#pragma unroll
    for (int e = 0; e < 16; ++e) {
      const int ka = e + 8 * half + ((e >= 8) ? 8 : 0);  // A: k = rel row
      const int kb = e + 8 * half + ((e >= 8) ? 8 : 0);  // B: k = rel row
      a.s[e] = qt[wid][ka * 16 + lr];
      b.s[e] = kt2[wid][kb * 16 + lr];
    }
    acc = __builtin_amdgcn_wmma_f32_16x16x32_bf16(false, a.v, false, b.v,
                                                  (short)0, acc, false, false);
    __syncthreads();
  }
  float* g = attnG + (size_t)bh * 256;
#pragma unroll
  for (int r = 0; r < 8; ++r)
    atomicAdd(&g[(r + 8 * half) * 16 + lr], acc[r]);
}

// Fold l2-norm scales + temperature, then row softmax over e (16 wide).
__global__ void attn_softmax(const float* __restrict__ attnG,
                             const float* __restrict__ nrm,
                             const float* __restrict__ temp,
                             float* __restrict__ attnS) {
  const int bh = blockIdx.x, img = bh >> 3, h = bh & 7;
  const int d = threadIdx.x >> 4, e = threadIdx.x & 15;
  __shared__ float sm[256];
  const float iq = 1.f / fmaxf(sqrtf(nrm[img * 256 + h * 16 + d]), 1e-12f);
  const float ik = 1.f / fmaxf(sqrtf(nrm[img * 256 + 128 + h * 16 + e]), 1e-12f);
  const float s = attnG[(size_t)bh * 256 + threadIdx.x] * iq * ik * temp[h];
  sm[threadIdx.x] = s;
  __syncthreads();
  float mx = -1e30f;
  for (int j = 0; j < 16; ++j) mx = fmaxf(mx, sm[d * 16 + j]);
  const float ex = expf(s - mx);
  __syncthreads();
  sm[threadIdx.x] = ex;
  __syncthreads();
  float sum = 0.f;
  for (int j = 0; j < 16; ++j) sum += sm[d * 16 + j];
  attnS[(size_t)bh * 256 + threadIdx.x] = ex / sum;
}

// out[px, h*16+d] = sum_e attn[b,h,d,e] * v[px, 256+h*16+e]
__global__ void attn_v(const unsigned short* __restrict__ qkv,
                       const float* __restrict__ attnS,
                       unsigned short* __restrict__ O1) {
  __shared__ float a_sh[2048];
  const int px0 = blockIdx.x * 2;
  const int img = px0 >> 14;
  for (int i = threadIdx.x; i < 2048; i += 256)
    a_sh[i] = attnS[(size_t)img * 2048 + i];
  __syncthreads();
  const int c  = threadIdx.x & 127;
  const int px = px0 + (threadIdx.x >> 7);
  const int h = c >> 4, d = c & 15;
  const float* ar = a_sh + h * 256 + d * 16;
  const unsigned short* vrow = qkv + (size_t)px * 384 + 256 + h * 16;
  float acc = 0.f;
#pragma unroll
  for (int e = 0; e < 16; ++e) acc += ar[e] * bf2f(vrow[e]);
  O1[(size_t)px * CCH + c] = f2bf(acc);
}

// LayerNorm over contiguous 128-float NHWC rows; one wave32 per pixel.
__global__ void ln2_k(const float* __restrict__ xa, const float* __restrict__ g,
                      const float* __restrict__ b_, unsigned short* __restrict__ A2) {
  const int wid = threadIdx.x >> 5, lane = threadIdx.x & 31;
  const int px = blockIdx.x * 8 + wid;
  const float4 v = *(const float4*)(xa + (size_t)px * CCH + lane * 4);
  float s  = v.x + v.y + v.z + v.w;
  float s2 = v.x * v.x + v.y * v.y + v.z * v.z + v.w * v.w;
  for (int o = 16; o > 0; o >>= 1) {
    s  += __shfl_xor(s, o, 32);
    s2 += __shfl_xor(s2, o, 32);
  }
  const float mu  = s * (1.f / 128.f);
  const float var = s2 * (1.f / 128.f) - mu * mu;
  const float rs  = rsqrtf(var + 1e-5f);
  unsigned short* o2 = A2 + (size_t)px * CCH + lane * 4;
  const float* gg = g + lane * 4;
  const float* bb = b_ + lane * 4;
  o2[0] = f2bf((v.x - mu) * rs * gg[0] + bb[0]);
  o2[1] = f2bf((v.y - mu) * rs * gg[1] + bb[1]);
  o2[2] = f2bf((v.z - mu) * rs * gg[2] + bb[2]);
  o2[3] = f2bf((v.w - mu) * rs * gg[3] + bb[3]);
}

// GDFN gating: g = gelu(u*c1); gated = [g*dw3(u), g*dw5(u)] -> bf16 [NPX,512]
__global__ void gdfn_gate(const unsigned short* __restrict__ U,
                          const float* __restrict__ c1, const float* __restrict__ c3,
                          const float* __restrict__ c5,
                          unsigned short* __restrict__ gated) {
  const int px = blockIdx.x, c = threadIdx.x;   // blockDim 256
  const int hh = (px >> 7) & 127, ww = px & 127;
  const float u  = bf2f(U[(size_t)px * 256 + c]);
  const float x1 = u * c1[c];
  const float g  = 0.5f * x1 * (1.f + erff(x1 * 0.70710678118654752f));
  float x3 = 0.f, x5 = 0.f;
  const float* w3 = c3 + c * 9;
  const float* w5 = c5 + c * 25;
  for (int dy = -2; dy <= 2; ++dy) {
    const int y = hh + dy; if ((unsigned)y >= 128u) continue;
    for (int dx = -2; dx <= 2; ++dx) {
      const int xx = ww + dx; if ((unsigned)xx >= 128u) continue;
      const float v = bf2f(U[(size_t)(px + dy * 128 + dx) * 256 + c]);
      x5 += v * w5[(dy + 2) * 5 + dx + 2];
      if (dy >= -1 && dy <= 1 && dx >= -1 && dx <= 1)
        x3 += v * w3[(dy + 1) * 3 + dx + 1];
    }
  }
  gated[(size_t)px * 512 + c]       = f2bf(g * x3);
  gated[(size_t)px * 512 + 256 + c] = f2bf(g * x5);
}

// ---------------------------------------------------------------------------
extern "C" void kernel_launch(void* const* d_in, const int* in_sizes, int n_in,
                              void* d_out, int out_size, void* d_ws, size_t ws_size,
                              hipStream_t stream) {
  (void)in_sizes; (void)n_in; (void)out_size; (void)ws_size;
  const float* x        = (const float*)d_in[0];
  const float* ln1_w    = (const float*)d_in[1];
  const float* ln1_b    = (const float*)d_in[2];
  const float* qkv_w    = (const float*)d_in[3];
  const float* qkv_dw   = (const float*)d_in[4];
  const float* temp     = (const float*)d_in[5];
  const float* attnproj = (const float*)d_in[6];
  const float* ln2_w    = (const float*)d_in[7];
  const float* ln2_b    = (const float*)d_in[8];
  const float* pin_w    = (const float*)d_in[9];
  const float* c1_w     = (const float*)d_in[10];
  const float* c3_w     = (const float*)d_in[11];
  const float* c5_w     = (const float*)d_in[12];
  const float* pout_w   = (const float*)d_in[13];

  char* ws = (char*)d_ws;
  unsigned short* qkvP  = (unsigned short*)(ws + 0);        // 384*128*2
  unsigned short* projP = (unsigned short*)(ws + 98304);    // 128*128*2
  unsigned short* pinP  = (unsigned short*)(ws + 131072);   // 256*128*2
  unsigned short* poutP = (unsigned short*)(ws + 196608);   // 128*512*2
  float* nrm   = (float*)(ws + 327680);                     // [8,256]
  float* attnG = (float*)(ws + 335872);                     // [64,256]
  float* attnS = (float*)(ws + 401408);                     // [64,256]
  const size_t o_xa   = (size_t)1 << 20;
  float* xa = (float*)(ws + o_xa);                          // NPX*128 f32
  const size_t o_big0 = o_xa + (size_t)NPX * 128 * 4;
  unsigned short* big0 = (unsigned short*)(ws + o_big0);    // Q1 / gated
  const size_t o_big1 = o_big0 + (size_t)NPX * 512 * 2;
  unsigned short* big1 = (unsigned short*)(ws + o_big1);    // qkv / U
  const size_t o_small = o_big1 + (size_t)NPX * 384 * 2;
  unsigned short* smallb = (unsigned short*)(ws + o_small); // A1 / O1 / A2

  // Weight packing into WMMA B-fragment layout + stat zeroing.
  pack_b<<<64, 256, 0, stream>>>(qkv_w,    qkvP,  384, 128);
  pack_b<<<64, 256, 0, stream>>>(attnproj, projP, 128, 128);
  pack_b<<<64, 256, 0, stream>>>(pin_w,    pinP,  256, 128);
  pack_b<<<64, 256, 0, stream>>>(pout_w,   poutP, 128, 512);
  zero_f32<<<(18432 + 255) / 256, 256, 0, stream>>>(nrm, 18432); // nrm+attnG

  // MDTA branch.
  ln1_k<<<NPX / 256, 256, 0, stream>>>(x, ln1_w, ln1_b, smallb);
  gemm_bf16_wmma<0><<<(8192 * 24) / 8, 256, 0, stream>>>(
      smallb, qkvP, NPX, 384, 128, big0, nullptr, nullptr);
  dwconv3_384<<<NPX, 384, 0, stream>>>(big0, qkv_dw, big1);
  qk_norm<<<NB * 16, 256, 0, stream>>>(big1, nrm);
  attn_qk_wmma<<<NB * NHEADS, 256, 0, stream>>>(big1, attnG);
  attn_softmax<<<NB * NHEADS, 256, 0, stream>>>(attnG, nrm, temp, attnS);
  attn_v<<<NPX / 2, 256, 0, stream>>>(big1, attnS, smallb);
  gemm_bf16_wmma<1><<<(8192 * 8) / 8, 256, 0, stream>>>(
      smallb, projP, NPX, 128, 128, nullptr, xa, x);      // xa = x + proj

  // GDFN branch.
  ln2_k<<<NPX / 8, 256, 0, stream>>>(xa, ln2_w, ln2_b, smallb);
  gemm_bf16_wmma<0><<<(8192 * 16) / 8, 256, 0, stream>>>(
      smallb, pinP, NPX, 256, 128, big1, nullptr, nullptr);
  gdfn_gate<<<NPX, 256, 0, stream>>>(big1, c1_w, c3_w, c5_w, big0);
  gemm_bf16_wmma<2><<<(8192 * 8) / 8, 256, 0, stream>>>(
      big0, poutP, NPX, 128, 512, nullptr, (float*)d_out, xa); // out = xa + pout
}
